// network_27728308863136
// MI455X (gfx1250) — matrix operand order repaired and checked by reading, not compiled, not run on hardware
//
#include <hip/hip_runtime.h>
#include <hip/hip_bf16.h>

// CDNA5 (gfx1250) persistent recurrent-scan kernel.
// 16 workgroups x 1024 threads (32 waves); each workgroup owns 16 batch rows
// for the whole T-step recurrence, each wave one 16-column H tile.
// Wh (bf16, 544KB on-chip) is split across BOTH on-chip pools:
//   - chunks 0..7  resident in VGPRs (64 regs/wave, fits the ~128 budget)
//   - chunks 8..16 resident in LDS (288KB), fragment-order layout
// inter=[x|h|cost] is bf16 in LDS; h accumulates in f32 WMMA accumulators.
// A and LDS-B fragments are double-buffered to overlap LDS latency with WMMA.

typedef __attribute__((ext_vector_type(16))) __bf16 v16bf;
typedef __attribute__((ext_vector_type(8)))  float  v8f;

#define HID    512
#define DD     529
#define BROWS  16            // batch rows per workgroup (WMMA M tile)
#define STR    548           // inter row stride in bf16 elems (>=544, bank spread)
#define KCH    17            // ceil(544/32) K chunks of 32
#define RES    8             // K chunks with VGPR-resident weights
#define LCH    (KCH - RES)   // K chunks with LDS-resident weights (9)
#define NW     32            // waves per workgroup
#define NTHR   1024

#define WLDS_BYTES  (LCH * NW * 32 * 32)          // 294912: [LCH][wave][lane][32B]
#define INTER_BYTES (BROWS * STR * 2)             // 17536
#define WO_BYTES    (DD * 4)                      // 2116
#define OPART_BYTES (BROWS * 32 * 4)              // 2048
#define SMEM_BYTES  (WLDS_BYTES + INTER_BYTES + WO_BYTES + OPART_BYTES)

__device__ __forceinline__ unsigned short f2bf(float f) {
  union { float f; unsigned u; } a; a.f = f;
  unsigned r = a.u + 0x7FFFu + ((a.u >> 16) & 1u);   // round-to-nearest-even
  return (unsigned short)(r >> 16);
}
__device__ __forceinline__ float bf2f(unsigned short s) {
  union { unsigned u; float f; } a; a.u = ((unsigned)s) << 16;
  return a.f;
}

union AFrag {
  v16bf v;
  unsigned long long q[4];
  unsigned short s[16];
};

__global__ __launch_bounds__(NTHR)
void rnn_scan_kernel(const float* __restrict__ x,      // [B, T, 16]
                     const float* __restrict__ hidden, // [B, H]
                     const float* __restrict__ cost,   // [B, T]
                     const float* __restrict__ Wo,     // [1, D]
                     const float* __restrict__ Wh,     // [H, D]
                     float* __restrict__ outs,         // [B, T]
                     float* __restrict__ hout,         // [B, H]
                     int T) {
  extern __shared__ __align__(16) unsigned char smem[];
  unsigned short* wLds   = (unsigned short*)smem;                          // weight frags
  unsigned short* interA = (unsigned short*)(smem + WLDS_BYTES);           // bf16 [16][STR]
  float*          WoS    = (float*)(smem + WLDS_BYTES + INTER_BYTES);
  float*          oPart  = (float*)(smem + WLDS_BYTES + INTER_BYTES + WO_BYTES);

  const int tid   = threadIdx.x;
  const int lane  = tid & 31;
  const int w     = tid >> 5;                // wave 0..31
  const int bbase = blockIdx.x * BROWS;

  const int nIdx     = lane & 15;            // N (or M-row) index within half
  const int nHalf    = lane >> 4;            // lane half select
  const int aHalfOff = nHalf * 8;            // A-matrix K offset / C-matrix M offset
  const int Cb       = w * 16;               // this wave's H-column tile

  // ---- Wo -> LDS (f32) ----
  for (int c = tid; c < DD; c += NTHR) WoS[c] = Wo[c];

  // ---- zero the K padding columns 529..547 (read by chunk 16) ----
  for (int e = tid; e < BROWS * (STR - DD); e += NTHR) {
    int r = e / (STR - DD), c = DD + e % (STR - DD);
    interA[r * STR + c] = 0;
  }

  // ---- initial inter: x(t=0) | h0 | cost(t=0), all bf16 ----
  for (int e = tid; e < BROWS * HID; e += NTHR) {
    int r = e >> 9, c = e & (HID - 1);
    interA[r * STR + 16 + c] = f2bf(hidden[(size_t)(bbase + r) * HID + c]);
  }
  if (tid < 256) {
    int r = tid >> 4, c = tid & 15;
    interA[r * STR + c] = f2bf(x[((size_t)(bbase + r) * T + 0) * 16 + c]);
  }
  if (tid < 16)
    interA[tid * STR + (DD - 1)] = f2bf(cost[(size_t)(bbase + tid) * T + 0]);

  // ---- weight B-fragments, bf16.
  // B-matrix lane layout (16x16x32): lane<16 -> N=lane, K=Kb..Kb+7,Kb+16..Kb+23 ; lane>=16 -> +8 K.
  // s[i] holds K = Kb + i + (i & 8).  Chunks 0..RES-1 -> VGPRs, chunks RES..16 -> LDS
  // (fragment order: [chunk][wave][lane] x 32B, so reads are 2 x b128/lane).
  v16bf wB[RES];
  {
    const float* r0 = Wh + (size_t)(Cb + nIdx) * DD;
#pragma unroll
    for (int k = 0; k < KCH; ++k) {
      const int Kb = k * 32 + nHalf * 8;
      AFrag f0;
#pragma unroll
      for (int i = 0; i < 16; ++i) {
        const int K = Kb + i + (i & 8);
        f0.s[i] = (K < DD) ? f2bf(r0[K]) : (unsigned short)0;
      }
      if (k < RES) {
        wB[k] = f0.v;
      } else {
        unsigned long long* wp =
            (unsigned long long*)(wLds + (((size_t)(k - RES) * NW + w) * 32 + lane) * 16);
        wp[0] = f0.q[0]; wp[1] = f0.q[1]; wp[2] = f0.q[2]; wp[3] = f0.q[3];
      }
    }
  }

  __syncthreads();

  // ---- one-step-ahead register prefetch of x / cost ----
  float xv = 0.f, cv = 0.f;
  if (T > 1) {
    if (tid < 256) {
      int r = tid >> 4, c = tid & 15;
      xv = x[((size_t)(bbase + r) * T + 1) * 16 + c];
    }
    if (tid < 16) cv = cost[(size_t)(bbase + tid) * T + 1];
  }

  const unsigned short* aBase = interA + nIdx * STR + aHalfOff;
  const unsigned short* bBase = wLds + ((size_t)w * 32 + lane) * 16;

  const v8f vzero = {};
  v8f acc = vzero;

  for (int t = 0; t < T; ++t) {
    // ---- h_{t+1} tile: 17 chained bf16 WMMAs; A (and LDS-B) double-buffered.
    acc = vzero;
    AFrag aCur, aNxt, bCur, bNxt;
    {
      const unsigned short* p = aBase;   // chunk 0
      aCur.q[0] = *(const unsigned long long*)(p + 0);
      aCur.q[1] = *(const unsigned long long*)(p + 4);
      aCur.q[2] = *(const unsigned long long*)(p + 16);
      aCur.q[3] = *(const unsigned long long*)(p + 20);
    }
#pragma unroll
    for (int k = 0; k < KCH; ++k) {
      if (k + 1 < KCH) {
        const unsigned short* p = aBase + (k + 1) * 32;
        aNxt.q[0] = *(const unsigned long long*)(p + 0);
        aNxt.q[1] = *(const unsigned long long*)(p + 4);
        aNxt.q[2] = *(const unsigned long long*)(p + 16);
        aNxt.q[3] = *(const unsigned long long*)(p + 20);
        if (k + 1 >= RES) {
          const unsigned long long* bp =
              (const unsigned long long*)(bBase + (size_t)(k + 1 - RES) * NW * 32 * 16);
          bNxt.q[0] = bp[0]; bNxt.q[1] = bp[1]; bNxt.q[2] = bp[2]; bNxt.q[3] = bp[3];
        }
      }
      const v16bf bop = (k < RES) ? wB[k] : bCur.v;
      acc = __builtin_amdgcn_wmma_f32_16x16x32_bf16(false, aCur.v, false, bop,
                                                    (short)0, acc, false, false);
      if (k + 1 < KCH) { aCur = aNxt; bCur = bNxt; }
    }

    // ---- o_t partial dots: 32 lanes per batch row sweep D (threads 0..511) ----
    if (tid < 512) {
      const int r  = tid & 15;
      const int cl = tid >> 4;   // 0..31
      float s = 0.f;
      for (int c = cl; c < DD; c += 32)
        s += bf2f(interA[r * STR + c]) * WoS[c];
      oPart[r * 32 + cl] = s;
    }
    __syncthreads();   // all reads of interA for step t complete

    // ---- write h_{t+1} back as bf16 (C layout: VGPR j -> M = j + 8*half) ----
#pragma unroll
    for (int j = 0; j < 8; ++j) {
      const int M = aHalfOff + j;
      interA[M * STR + 16 + Cb + nIdx] = f2bf(acc[j]);
    }

    // ---- reduce + store o_t ----
    if (tid < 16) {
      float s = 0.f;
#pragma unroll
      for (int i = 0; i < 32; ++i) s += oPart[tid * 32 + i];
      outs[(size_t)(bbase + tid) * T + t] = s;
    }

    // ---- stage prefetched x_{t+1}/cost_{t+1}; issue loads for t+2 ----
    if (t + 1 < T) {
      if (tid < 256) {
        int r = tid >> 4, c = tid & 15;
        interA[r * STR + c] = f2bf(xv);
        if (t + 2 < T)
          xv = x[((size_t)(bbase + r) * T + (t + 2)) * 16 + c];
      }
      if (tid < 16) {
        interA[tid * STR + (DD - 1)] = f2bf(cv);
        if (t + 2 < T)
          cv = cost[(size_t)(bbase + tid) * T + (t + 2)];
      }
    }
    __syncthreads();   // interA now holds step t+1 inputs
  }

  // ---- final hidden state, straight from last-step f32 accumulators ----
#pragma unroll
  for (int j = 0; j < 8; ++j) {
    const int M = aHalfOff + j;
    hout[(size_t)(bbase + M) * HID + Cb + nIdx] = acc[j];
  }
}

extern "C" void kernel_launch(void* const* d_in, const int* in_sizes, int n_in,
                              void* d_out, int out_size, void* d_ws, size_t ws_size,
                              hipStream_t stream) {
  const float* x      = (const float*)d_in[0];
  const float* hidden = (const float*)d_in[1];
  const float* cost   = (const float*)d_in[2];
  const float* Wo     = (const float*)d_in[3];
  const float* Wh     = (const float*)d_in[4];

  const int B = in_sizes[1] / HID;   // 256
  const int T = in_sizes[2] / B;     // 2048

  float* outs = (float*)d_out;               // [B, T]
  float* hout = outs + (size_t)B * T;        // [B, H]

  (void)hipFuncSetAttribute((const void*)rnn_scan_kernel,
                            hipFuncAttributeMaxDynamicSharedMemorySize,
                            (int)SMEM_BYTES);

  rnn_scan_kernel<<<B / BROWS, NTHR, SMEM_BYTES, stream>>>(x, hidden, cost, Wo, Wh,
                                                           outs, hout, T);
}
